// DiagonalSSMLayer_62302795596619
// MI455X (gfx1250) — matrix operand order repaired and checked by reading, not compiled, not run on hardware
//
#include <hip/hip_runtime.h>
#include <stdint.h>
#include <stddef.h>

// ---------------------------------------------------------------------------
// DiagonalSSMLayer for MI455X (gfx1250, wave32, WMMA).
// B=8, S=8192, D=1024, Hn=32, T=B*S=65536 tokens.
// Memory-bound problem (~0.8 GB traffic, ~13 GFLOP): fuse LN + input
// projections (one read of x), chunked associative scan in L2-resident
// [k][T] layout, fused output projection + residual (second read of x).
// GEMMs use v_wmma_f32_16x16x32_bf16 (fp32 accumulate).
// ---------------------------------------------------------------------------

typedef __attribute__((ext_vector_type(16))) __bf16 v16bf;
typedef __attribute__((ext_vector_type(8)))  float  v8f;

#define D_DIM 1024
#define HN    32
#define S_LEN 8192
#define CHUNK 512
#define NCHUNK (S_LEN / CHUNK)   // 16
#define NSEQ   256               // B * HN = 8 * 32

__device__ __forceinline__ uint32_t f2bf(float f) {
    uint32_t u = __float_as_uint(f);
    // round-to-nearest-even truncation to bf16
    return (u + 0x7FFFu + ((u >> 16) & 1u)) >> 16;
}
__device__ __forceinline__ uint32_t packbf2(float lo, float hi) {
    return f2bf(lo) | (f2bf(hi) << 16);
}

union FragAB { v16bf v; uint4 q[2]; uint32_t u[8]; };

// ---------------------------------------------------------------------------
// Kernel 0: prepack weights into per-lane WMMA B-fragment order (bf16 pairs).
// Wcat = [W_a ; W_in] as a (64 x 1024) matrix -> layout [nt(4)][ks(32)][lane][8 dw]
// W_out^T (32 x 1024)                         -> layout [nt(64)][lane][8 dw]
// B-matrix element i of a lane <-> K = ks*32 + 8*g + (i&7) + 16*(i>>3)  (g = lane>>4, n = nt*16 + lane%16)
// ---------------------------------------------------------------------------
__global__ void prepack_kernel(const float* __restrict__ W_a,
                               const float* __restrict__ W_in,
                               const float* __restrict__ W_out,
                               uint32_t* __restrict__ Wcat_pk,
                               uint32_t* __restrict__ Wout_pk)
{
    int idx = blockIdx.x * blockDim.x + threadIdx.x;
    const int NCAT = 4 * 32 * 32 * 8;   // 32768 dwords
    if (idx < NCAT) {
        int vd = idx & 7, lane = (idx >> 3) & 31, ks = (idx >> 8) & 31, nt = idx >> 13;
        int n  = nt * 16 + (lane & 15);
        int g  = lane >> 4;
        int k0 = ks * 32 + 8 * g + ((vd >> 2) << 4) + ((vd & 3) << 1);
        const float* row = (n < 32) ? (W_a + (size_t)n * D_DIM)
                                    : (W_in + (size_t)(n - 32) * D_DIM);
        Wcat_pk[idx] = packbf2(row[k0], row[k0 + 1]);
    } else {
        int e = idx - NCAT;
        if (e < 64 * 32 * 8) {
            int vd = e & 7, lane = (e >> 3) & 31, nt = e >> 8;
            int dcol = nt * 16 + (lane & 15);
            int g  = lane >> 4;
            int k0 = 8 * g + ((vd >> 2) << 4) + ((vd & 3) << 1);
            // B element (k, dcol) = W_out[dcol][k], W_out row-major (1024 x 32)
            Wout_pk[e] = packbf2(W_out[(size_t)dcol * HN + k0],
                                 W_out[(size_t)dcol * HN + k0 + 1]);
        }
    }
}

// ---------------------------------------------------------------------------
// Kernel 1: fused LayerNorm + alpha/b projections for 16 tokens per block.
// 128 threads = 4 waves. LN with 8 threads/token, xn stored as bf16 in LDS,
// then each wave computes one 16x16 output tile via 32 WMMA k-steps (K=1024).
// Outputs written in [k][T] layout: alpha (post-sigmoid) and b (pre-scan).
// ---------------------------------------------------------------------------
__launch_bounds__(128)
__global__ void ln_proj_kernel(const float* __restrict__ x,
                               const float* __restrict__ b_a,
                               const float* __restrict__ b_in,
                               const float* __restrict__ ln_g,
                               const float* __restrict__ ln_b,
                               const uint32_t* __restrict__ Wcat_pk,
                               float* __restrict__ alpha_arr,
                               float* __restrict__ b_arr,
                               int T)
{
    __shared__ __align__(16) uint32_t xn_u[16 * 512];  // 16 tokens x 1024 bf16 (32 KB)
    __shared__ float red_s[128], red_q[128];
    __shared__ float sh_mu[16], sh_rs[16];

    const int tile = blockIdx.x;
    const int tid  = threadIdx.x;
    const int tok  = tid >> 3;        // 16 tokens, 8 threads each
    const int part = tid & 7;
    const float4* xr = (const float4*)(x + (size_t)(tile * 16 + tok) * D_DIM);

    // pass 1: sum / sumsq (32 float4 per thread, coalesced in groups of 8)
    float s = 0.f, s2 = 0.f;
    #pragma unroll 4
    for (int i = 0; i < 32; ++i) {
        float4 v = xr[part + 8 * i];
        s  += v.x + v.y + v.z + v.w;
        s2 += v.x * v.x + v.y * v.y + v.z * v.z + v.w * v.w;
    }
    red_s[tid] = s; red_q[tid] = s2;
    __syncthreads();
    if (part == 0) {
        float a = 0.f, b = 0.f;
        #pragma unroll
        for (int j = 0; j < 8; ++j) { a += red_s[tok * 8 + j]; b += red_q[tok * 8 + j]; }
        float mu  = a * (1.0f / D_DIM);
        float var = b * (1.0f / D_DIM) - mu * mu;
        sh_mu[tok] = mu;
        sh_rs[tok] = rsqrtf(var + 1e-5f);
    }
    __syncthreads();

    // pass 2: normalize, affine, convert to bf16 into LDS (re-reads hit WGP$)
    {
        const float mu = sh_mu[tok], rs = sh_rs[tok];
        const float4* gp = (const float4*)ln_g;
        const float4* bp = (const float4*)ln_b;
        #pragma unroll 4
        for (int i = 0; i < 32; ++i) {
            int f4 = part + 8 * i;
            float4 v  = xr[f4];
            float4 gm = gp[f4];
            float4 bt = bp[f4];
            float e0 = (v.x - mu) * rs * gm.x + bt.x;
            float e1 = (v.y - mu) * rs * gm.y + bt.y;
            float e2 = (v.z - mu) * rs * gm.z + bt.z;
            float e3 = (v.w - mu) * rs * gm.w + bt.w;
            xn_u[tok * 512 + f4 * 2]     = packbf2(e0, e1);
            xn_u[tok * 512 + f4 * 2 + 1] = packbf2(e2, e3);
        }
    }
    __syncthreads();

    // WMMA: wave w computes n-tile w (outputs w*16 .. w*16+15 of the 64 concat outputs)
    const int wave = tid >> 5, lane = tid & 31;
    const int g = lane >> 4, mrow = lane & 15;
    v8f acc = {0.f, 0.f, 0.f, 0.f, 0.f, 0.f, 0.f, 0.f};
    const uint32_t* wb   = Wcat_pk + ((size_t)(wave * 32) * 32 + lane) * 8;
    const uint32_t* arow = xn_u + mrow * 512;
    for (int ks = 0; ks < 32; ++ks) {
        FragAB fa, fb;
        // A element i <-> K = ks*32 + 8g + (i&7) + 16*(i>>3): two b128 LDS loads
        int kb = ks * 32 + 8 * g;
        fa.q[0] = *(const uint4*)(arow + (kb >> 1));
        fa.q[1] = *(const uint4*)(arow + ((kb + 16) >> 1));
        const uint32_t* bsrc = wb + (size_t)ks * (32 * 8);
        fb.q[0] = *(const uint4*)(bsrc);
        fb.q[1] = *(const uint4*)(bsrc + 4);
        acc = __builtin_amdgcn_wmma_f32_16x16x32_bf16(false, fa.v, false, fb.v,
                                                      (short)0, acc, false, false);
    }

    // epilogue: C vgpr r <-> (m = r + 8g, n = lane%16); write [k][T] layout
    const int nl = lane & 15;
    #pragma unroll
    for (int r = 0; r < 8; ++r) {
        int m = r + 8 * g;
        int t = tile * 16 + m;
        float c = acc[r];
        if (wave < 2) {
            int k = wave * 16 + nl;
            float vlog = c + b_a[k];
            alpha_arr[(size_t)k * T + t] = 1.0f / (1.0f + __expf(-vlog));
        } else {
            int k = (wave - 2) * 16 + nl;
            b_arr[(size_t)k * T + t] = c + b_in[k];
        }
    }
}

// ---------------------------------------------------------------------------
// Kernels 2-4: chunked linear-recurrence scan h_t = a_t*h_{t-1} + b_t along S.
// 256 sequences (k,b) x 16 chunks of 512. Data is 16 MB total -> L2 resident.
// ---------------------------------------------------------------------------
__global__ void scan_partial_kernel(const float* __restrict__ alpha_arr,
                                    const float* __restrict__ b_arr,
                                    float* __restrict__ cA, float* __restrict__ cU, int T)
{
    int idx = blockIdx.x * blockDim.x + threadIdx.x;
    if (idx >= NSEQ * NCHUNK) return;
    int seq = idx >> 4;           // 0..255
    int c   = idx & (NCHUNK - 1);
    int k = seq >> 3, bb = seq & 7;
    size_t base = (size_t)k * T + (size_t)bb * S_LEN + (size_t)c * CHUNK;
    float P = 1.f, U = 0.f;
    for (int s = 0; s < CHUNK; ++s) {
        float a  = alpha_arr[base + s];
        float bv = b_arr[base + s];
        U = fmaf(a, U, bv);
        P *= a;
    }
    cA[idx] = P; cU[idx] = U;
}

__global__ void scan_carry_kernel(const float* __restrict__ cA,
                                  const float* __restrict__ cU,
                                  float* __restrict__ carry)
{
    int seq = blockIdx.x * blockDim.x + threadIdx.x;
    if (seq >= NSEQ) return;
    float h = 0.f;
    for (int c = 0; c < NCHUNK; ++c) {
        int i = seq * NCHUNK + c;
        carry[i] = h;
        h = fmaf(cA[i], h, cU[i]);
    }
}

__global__ void scan_apply_kernel(const float* __restrict__ alpha_arr,
                                  float* __restrict__ bh,       // b in, h out (in place)
                                  const float* __restrict__ carry, int T)
{
    int idx = blockIdx.x * blockDim.x + threadIdx.x;
    if (idx >= NSEQ * NCHUNK) return;
    int seq = idx >> 4;
    int c   = idx & (NCHUNK - 1);
    int k = seq >> 3, bb = seq & 7;
    size_t base = (size_t)k * T + (size_t)bb * S_LEN + (size_t)c * CHUNK;
    float h = carry[idx];
    for (int s = 0; s < CHUNK; ++s) {
        float a  = alpha_arr[base + s];
        float bv = bh[base + s];
        h = fmaf(a, h, bv);
        bh[base + s] = h;
    }
}

// ---------------------------------------------------------------------------
// Kernel 5: out = x + h @ W_out^T + b_out. 16 tokens/block, 8 waves; each
// wave handles 8 of the 64 d-column n-tiles, one K=32 WMMA per tile.
// ---------------------------------------------------------------------------
__launch_bounds__(256)
__global__ void out_proj_kernel(const float* __restrict__ x,
                                const float* __restrict__ h_arr,   // [k][T]
                                const uint32_t* __restrict__ Wout_pk,
                                const float* __restrict__ b_out,
                                float* __restrict__ out, int T)
{
    const int tile = blockIdx.x;
    const int tid  = threadIdx.x;
    const int wave = tid >> 5, lane = tid & 31;
    const int g = lane >> 4, mrow = lane & 15;
    const int t = tile * 16 + mrow;

    // A fragment: h[t][k], element pair v: K0 = 16*(v>>2) + 8g + 2*(v&3)
    FragAB fa;
    #pragma unroll
    for (int v = 0; v < 8; ++v) {
        int k0 = ((v >> 2) << 4) + 8 * g + ((v & 3) << 1);
        float f0 = h_arr[(size_t)k0 * T + t];
        float f1 = h_arr[(size_t)(k0 + 1) * T + t];
        fa.u[v] = packbf2(f0, f1);
    }

    const int nl = lane & 15;
    for (int j = 0; j < 8; ++j) {
        int nt = wave * 8 + j;                     // d-columns nt*16 .. nt*16+15
        FragAB fb;
        const uint32_t* bsrc = Wout_pk + ((size_t)nt * 32 + lane) * 8;
        fb.q[0] = *(const uint4*)bsrc;
        fb.q[1] = *(const uint4*)(bsrc + 4);
        v8f acc = {0.f, 0.f, 0.f, 0.f, 0.f, 0.f, 0.f, 0.f};
        acc = __builtin_amdgcn_wmma_f32_16x16x32_bf16(false, fa.v, false, fb.v,
                                                      (short)0, acc, false, false);
        #pragma unroll
        for (int r = 0; r < 8; ++r) {
            int m  = r + 8 * g;
            int tt = tile * 16 + m;
            int dcol = nt * 16 + nl;
            size_t idx = (size_t)tt * D_DIM + dcol;
            out[idx] = acc[r] + b_out[dcol] + x[idx];
        }
    }
}

// ---------------------------------------------------------------------------
// Launch. Inputs (setup_inputs order): x, W_a, b_a, W_in, b_in, W_out, b_out,
// ln_gamma, ln_beta. Workspace usage ~16.8 MB.
// ---------------------------------------------------------------------------
extern "C" void kernel_launch(void* const* d_in, const int* in_sizes, int n_in,
                              void* d_out, int out_size, void* d_ws, size_t ws_size,
                              hipStream_t stream)
{
    (void)n_in; (void)out_size; (void)ws_size;
    const float* x     = (const float*)d_in[0];
    const float* W_a   = (const float*)d_in[1];
    const float* b_a   = (const float*)d_in[2];
    const float* W_in  = (const float*)d_in[3];
    const float* b_in  = (const float*)d_in[4];
    const float* W_out = (const float*)d_in[5];
    const float* b_out = (const float*)d_in[6];
    const float* ln_g  = (const float*)d_in[7];
    const float* ln_b  = (const float*)d_in[8];
    float* out = (float*)d_out;

    const int T = in_sizes[0] / D_DIM;   // 65536 tokens

    char* ws = (char*)d_ws;
    size_t off = 0;
    float*    alpha_arr = (float*)(ws + off);    off += (size_t)HN * T * 4;        // 8 MB
    float*    bh_arr    = (float*)(ws + off);    off += (size_t)HN * T * 4;        // 8 MB
    uint32_t* Wcat_pk   = (uint32_t*)(ws + off); off += (size_t)4 * 32 * 32 * 8 * 4;
    uint32_t* Wout_pk   = (uint32_t*)(ws + off); off += (size_t)64 * 32 * 8 * 4;
    float*    cA        = (float*)(ws + off);    off += (size_t)NSEQ * NCHUNK * 4;
    float*    cU        = (float*)(ws + off);    off += (size_t)NSEQ * NCHUNK * 4;
    float*    carry     = (float*)(ws + off);    off += (size_t)NSEQ * NCHUNK * 4;

    const int ntiles = T / 16;           // 4096 token tiles
    const int npack  = 4 * 32 * 32 * 8 + 64 * 32 * 8;   // 49152 dwords

    prepack_kernel<<<(npack + 255) / 256, 256, 0, stream>>>(W_a, W_in, W_out, Wcat_pk, Wout_pk);
    ln_proj_kernel<<<ntiles, 128, 0, stream>>>(x, b_a, b_in, ln_g, ln_b, Wcat_pk,
                                               alpha_arr, bh_arr, T);
    scan_partial_kernel<<<(NSEQ * NCHUNK + 255) / 256, 256, 0, stream>>>(alpha_arr, bh_arr,
                                                                         cA, cU, T);
    scan_carry_kernel<<<1, 256, 0, stream>>>(cA, cU, carry);
    scan_apply_kernel<<<(NSEQ * NCHUNK + 255) / 256, 256, 0, stream>>>(alpha_arr, bh_arr,
                                                                       carry, T);
    out_proj_kernel<<<ntiles, 256, 0, stream>>>(x, bh_arr, Wout_pk, b_out, out, T);
}
